// NetVLAD_65755949302226
// MI455X (gfx1250) — matrix operand order repaired and checked by reading, not compiled, not run on hardware
//
#include <hip/hip_runtime.h>
#include <hip/hip_bf16.h>

typedef __bf16 bf16;
typedef __attribute__((ext_vector_type(8)))  bf16  v8bf;
typedef __attribute__((ext_vector_type(16))) bf16  v16bf;
typedef __attribute__((ext_vector_type(8)))  float v8f;

#define B_      32
#define N_      8192
#define D_      256
#define K_      64
#define NCHUNK  512
#define SUBT    32
#define CHUNKS  (N_ / NCHUNK)

// LDS row strides (elements). All chosen so byte stride is a multiple of 16
// (b128 ds_load alignment) and not a multiple of 64 banks.
#define WS_STRIDE 264   // 528 B
#define XS_STRIDE 264   // 528 B
#define XT_STRIDE 40    // 80 B
#define AT_STRIDE 40    // 80 B
#define LG_STRIDE 68    // 272 B

// Build a v16bf fragment from two 16-byte LDS chunks (A-matrix 16-bit layout:
// per lane, VGPR0-3 hold K = half*8..+7, VGPR4-7 hold K = 16+half*8..+7).
__device__ inline v16bf frag_split(const bf16* p, int lo_off, int hi_off) {
    v8bf lo = *(const v8bf*)(p + lo_off);
    v8bf hi = *(const v8bf*)(p + hi_off);
    return __builtin_shufflevector(lo, hi, 0,1,2,3,4,5,6,7,8,9,10,11,12,13,14,15);
}

// Contiguous k-order fragment (unused here): per lane 16 contiguous K values from p.
__device__ inline v16bf frag_contig(const bf16* p) {
    v8bf lo = *(const v8bf*)(p);
    v8bf hi = *(const v8bf*)(p + 8);
    return __builtin_shufflevector(lo, hi, 0,1,2,3,4,5,6,7,8,9,10,11,12,13,14,15);
}

__global__ void netvlad_zero_ws(float* __restrict__ p, int n) {
    int i = blockIdx.x * 256 + threadIdx.x;
    if (i < n) p[i] = 0.0f;
}

// Fused pass: logits -> softmax -> agg accumulation, reading x exactly once.
// grid = B * CHUNKS blocks, 256 threads (8 waves of 32).
__global__ __launch_bounds__(256)
void netvlad_main(const float* __restrict__ x, const float* __restrict__ W,
                  const float* __restrict__ bias,
                  float* __restrict__ agg, float* __restrict__ asum_g) {
    __shared__ bf16  Ws [K_][WS_STRIDE];     // W as bf16, row-major [kc][d]
    __shared__ bf16  xs [SUBT][XS_STRIDE];   // x tile row-major   [n][d]
    __shared__ bf16  xts[D_][XT_STRIDE];     // x tile transposed  [d][n]
    __shared__ float lg [SUBT][LG_STRIDE];   // logits tile        [n][kc]
    __shared__ bf16  ats[K_][AT_STRIDE];     // a transposed       [kc][n]
    __shared__ float asum_s[K_];

    const int t    = threadIdx.x;
    const int lane = t & 31;
    const int wave = t >> 5;
    const int l16  = lane & 15;
    const int half = lane >> 4;

    const int b     = blockIdx.x / CHUNKS;
    const int chunk = blockIdx.x % CHUNKS;
    const float* xg = x + ((size_t)b * N_ + (size_t)chunk * NCHUNK) * D_;

    // Stage W (f32 -> bf16) into LDS, vectorized.
    for (int i4 = t; i4 < (K_ * D_) / 4; i4 += 256) {
        float4 v = ((const float4*)W)[i4];
        int base = i4 * 4;
        int kc = base >> 8, d = base & 255;
        Ws[kc][d + 0] = (bf16)v.x; Ws[kc][d + 1] = (bf16)v.y;
        Ws[kc][d + 2] = (bf16)v.z; Ws[kc][d + 3] = (bf16)v.w;
    }
    if (t < K_) asum_s[t] = 0.0f;

    // Wave roles.
    const int mtile = wave >> 2, kct  = wave & 3;   // GEMM1: 16x16 logits tile
    const int krow  = wave & 3,  dgrp = wave >> 2;  // GEMM2: kc row, 8 d-tiles
    const float bv = bias[kct * 16 + l16];          // bias for this lane's column

    v8f aggacc[8];
    #pragma unroll
    for (int i = 0; i < 8; ++i) aggacc[i] = (v8f){0,0,0,0,0,0,0,0};
    float asum_loc[8];
    #pragma unroll
    for (int j = 0; j < 8; ++j) asum_loc[j] = 0.0f;

    __syncthreads();

    for (int sub = 0; sub < NCHUNK / SUBT; ++sub) {
        const float* xsub = xg + (size_t)sub * SUBT * D_;
        if (sub + 1 < NCHUNK / SUBT)
            __builtin_prefetch(xsub + SUBT * D_ + t * 32, 0, 3);  // global_prefetch_b8

        // Stage 32x256 f32 tile -> bf16 (row-major + transposed), float4 loads.
        #pragma unroll
        for (int i = 0; i < 8; ++i) {
            int f  = t + i * 256;
            int r  = f >> 6;
            int c4 = (f & 63) << 2;
            float4 v = *(const float4*)(xsub + r * D_ + c4);
            bf16 b0 = (bf16)v.x, b1 = (bf16)v.y, b2 = (bf16)v.z, b3 = (bf16)v.w;
            xs[r][c4 + 0] = b0; xs[r][c4 + 1] = b1;
            xs[r][c4 + 2] = b2; xs[r][c4 + 3] = b3;
            xts[c4 + 0][r] = b0; xts[c4 + 1][r] = b1;
            xts[c4 + 2][r] = b2; xts[c4 + 3][r] = b3;
        }
        __syncthreads();

        // GEMM1: logits(32x64) = x_tile(32x256) * W^T. One 16x16 tile per wave.
        {
            v8f acc = (v8f){0,0,0,0,0,0,0,0};
            const int m  = mtile * 16 + l16;  // x row
            const int nc = kct * 16 + l16;    // W row (= output column)
            #pragma unroll
            for (int dk = 0; dk < D_; dk += 32) {
                v16bf af = frag_split(&xs[m][dk], half * 8, 16 + half * 8);
                v16bf bf_ = frag_split(&Ws[nc][dk], half * 8, 16 + half * 8);
                acc = __builtin_amdgcn_wmma_f32_16x16x32_bf16(
                          false, af, false, bf_, (short)0, acc, false, false);
            }
            #pragma unroll
            for (int j = 0; j < 8; ++j)
                lg[mtile * 16 + half * 8 + j][kct * 16 + l16] = acc[j] + bv;
        }
        __syncthreads();

        // Softmax over kc: 8 lanes per row, shfl_xor reductions (wave32).
        {
            const int r   = t >> 3;
            const int kcb = (t & 7) << 3;
            float v[8], mx = -3.402823466e38f;
            #pragma unroll
            for (int j = 0; j < 8; ++j) { v[j] = lg[r][kcb + j]; mx = fmaxf(mx, v[j]); }
            mx = fmaxf(mx, __shfl_xor(mx, 1, 32));
            mx = fmaxf(mx, __shfl_xor(mx, 2, 32));
            mx = fmaxf(mx, __shfl_xor(mx, 4, 32));
            float s = 0.0f;
            #pragma unroll
            for (int j = 0; j < 8; ++j) { v[j] = __expf(v[j] - mx); s += v[j]; }
            s += __shfl_xor(s, 1, 32);
            s += __shfl_xor(s, 2, 32);
            s += __shfl_xor(s, 4, 32);
            const float inv = 1.0f / s;
            #pragma unroll
            for (int j = 0; j < 8; ++j) {
                float a = v[j] * inv;
                asum_loc[j] += a;
                ats[kcb + j][r] = (bf16)a;
            }
        }
        __syncthreads();

        // GEMM2: agg(64x256) += a^T(64x32) * x_tile(32x256). 8 tiles per wave.
        {
            v16bf af = frag_split(&ats[krow * 16 + l16][0], half * 8, 16 + half * 8);
            #pragma unroll
            for (int i = 0; i < 8; ++i) {
                const int d = (dgrp * 8 + i) * 16 + l16;
                v16bf bf_ = frag_split(&xts[d][0], half * 8, 16 + half * 8);
                aggacc[i] = __builtin_amdgcn_wmma_f32_16x16x32_bf16(
                                false, af, false, bf_, (short)0, aggacc[i], false, false);
            }
        }
        __syncthreads();  // before next subtile overwrites xs/xts
    }

    // Flush a_sum: LDS reduce, then one global atomic per kc.
    {
        const int kcb = (t & 7) << 3;
        #pragma unroll
        for (int j = 0; j < 8; ++j) atomicAdd(&asum_s[kcb + j], asum_loc[j]);
    }
    // Flush agg accumulators: f32 global atomics (2 MB total target, tiny traffic).
    {
        float* aggb = agg + (size_t)b * K_ * D_;
        #pragma unroll
        for (int i = 0; i < 8; ++i) {
            const int d = (dgrp * 8 + i) * 16 + l16;
            #pragma unroll
            for (int j = 0; j < 8; ++j) {
                const int kc = krow * 16 + half * 8 + j;
                atomicAdd(&aggb[kc * D_ + d], aggacc[i][j]);
            }
        }
    }
    __syncthreads();
    if (t < K_) atomicAdd(&asum_g[b * K_ + t], asum_s[t]);
}

// Finalize: vlad = agg - a_sum*c, L2-normalize over D. One block per (b,kc).
__global__ __launch_bounds__(256)
void netvlad_finalize(const float* __restrict__ agg, const float* __restrict__ asum_g,
                      const float* __restrict__ cent, float* __restrict__ out) {
    __shared__ float red[8];
    const int b  = blockIdx.x >> 6;
    const int kc = blockIdx.x & 63;
    const int d  = threadIdx.x;
    const int lane = threadIdx.x & 31, wave = threadIdx.x >> 5;

    const float s = asum_g[b * K_ + kc];
    const float v = agg[((size_t)b * K_ + kc) * D_ + d] - s * cent[kc * D_ + d];

    float sq = v * v;
    #pragma unroll
    for (int off = 16; off > 0; off >>= 1) sq += __shfl_xor(sq, off, 32);
    if (lane == 0) red[wave] = sq;
    __syncthreads();
    if (wave == 0) {
        float r = (lane < 8) ? red[lane] : 0.0f;
        r += __shfl_xor(r, 4, 32);
        r += __shfl_xor(r, 2, 32);
        r += __shfl_xor(r, 1, 32);
        if (lane == 0) red[0] = r;
    }
    __syncthreads();
    const float nrm = sqrtf(red[0]);
    const float inv = 1.0f / fmaxf(nrm, 1e-12f);
    out[((size_t)b * K_ + kc) * D_ + d] = v * inv;
}

extern "C" void kernel_launch(void* const* d_in, const int* in_sizes, int n_in,
                              void* d_out, int out_size, void* d_ws, size_t ws_size,
                              hipStream_t stream) {
    const float* x    = (const float*)d_in[0];   // [B,N,D]
    const float* W    = (const float*)d_in[1];   // [K,D]
    const float* bias = (const float*)d_in[2];   // [K]
    const float* cent = (const float*)d_in[3];   // [K,D]
    float* out = (float*)d_out;                  // [B, K*D]

    float* agg  = (float*)d_ws;                  // [B,K,D] f32
    float* asum = agg + (size_t)B_ * K_ * D_;    // [B,K]   f32
    const int ztot = B_ * K_ * D_ + B_ * K_;

    netvlad_zero_ws<<<(ztot + 255) / 256, 256, 0, stream>>>(agg, ztot);
    netvlad_main<<<B_ * CHUNKS, 256, 0, stream>>>(x, W, bias, agg, asum);
    netvlad_finalize<<<B_ * K_, 256, 0, stream>>>(agg, asum, cent, out);
}
